// GraphAttentionLayer_83811991814212
// MI455X (gfx1250) — matrix operand order repaired and checked by reading, not compiled, not run on hardware
//
#include <hip/hip_runtime.h>
#include <hip/hip_bf16.h>
#include <stdint.h>

// B=32, N=4096, F_IN=F_OUT=64, ALPHA=0.2
// m = b*64+d (M=2048), j = K (4096, j=4095 masked to zero), i = N (4096)

typedef __bf16 bf16;
typedef __attribute__((ext_vector_type(16))) __bf16 v16bf;
typedef __attribute__((ext_vector_type(8)))  __bf16 v8bf;
typedef __attribute__((ext_vector_type(8)))  float  v8f;

#define NN   4096
#define BB   32
#define FD   64
#define MDIM (BB*FD)   // 2048

// ---------------------------------------------------------------------------
// Kernel 1: transpose adj -> bf16 mask  MT[i][j] = (j<4095 && adj[j+1][i]>0)
// ---------------------------------------------------------------------------
__global__ void __launch_bounds__(256) prep_mask(const int* __restrict__ adj,
                                                 bf16* __restrict__ MT) {
  __shared__ bf16 tile[64][64 + 2];
  const int i0 = blockIdx.x * 64;
  const int j0 = blockIdx.y * 64;
  const int t  = threadIdx.x;
#pragma unroll
  for (int p = 0; p < 16; ++p) {
    int idx = t + p * 256;           // 0..4095
    int jj = idx >> 6, ii = idx & 63;
    int gj = j0 + jj;
    float v = 0.0f;
    if (gj < NN - 1) v = (adj[(size_t)(gj + 1) * NN + (i0 + ii)] > 0) ? 1.0f : 0.0f;
    tile[ii][jj] = (bf16)v;          // transpose in LDS
  }
  __syncthreads();
#pragma unroll
  for (int p = 0; p < 16; ++p) {
    int idx = t + p * 256;
    int ii = idx >> 6, jj = idx & 63;
    MT[(size_t)(i0 + ii) * NN + (j0 + jj)] = tile[ii][jj];
  }
}

// ---------------------------------------------------------------------------
// Kernel 2: h = inp @ W (row 0 zeroed), write Hbd[bd][j] (bf16) and
//           score1[b,j] = sum_d h[b,j,d] * a1[d,j]
// ---------------------------------------------------------------------------
__global__ void __launch_bounds__(256) prep_h(const float* __restrict__ inp,
                                              const float* __restrict__ W,
                                              const float* __restrict__ a,
                                              bf16*  __restrict__ Hbd,
                                              float* __restrict__ score1) {
  __shared__ float sIn[64][65];   // [j][f], padded
  __shared__ float sW[64][64];    // [f][d]
  __shared__ float sRed[256];
  const int b  = blockIdx.y;
  const int j0 = blockIdx.x * 64;
  const int t  = threadIdx.x;
#pragma unroll
  for (int p = 0; p < 16; ++p) {
    int idx = t + p * 256;
    int jj = idx >> 6, f = idx & 63;
    sIn[jj][f] = inp[((size_t)b * NN + j0 + jj) * FD + f];
  }
#pragma unroll
  for (int p = 0; p < 16; ++p) {
    int idx = t + p * 256;
    int f = idx >> 6, d = idx & 63;
    sW[f][d] = W[f * FD + d];
  }
  __syncthreads();
  const int j  = t & 63;       // row within tile
  const int dq = t >> 6;       // 0..3
  const int gj = j0 + j;
  float part = 0.0f;
#pragma unroll
  for (int k = 0; k < 16; ++k) {
    int d = dq + 4 * k;
    float acc = 0.0f;
#pragma unroll
    for (int f = 0; f < 64; ++f) acc += sIn[j][f] * sW[f][d];
    if (gj == 0) acc = 0.0f;
    Hbd[(size_t)(b * FD + d) * NN + gj] = (bf16)acc;      // coalesced over j
    part += acc * a[(size_t)d * NN + gj];                 // a1[d][i]
  }
  sRed[t] = part;
  __syncthreads();
  if (t < 64) {
    float s = sRed[t] + sRed[t + 64] + sRed[t + 128] + sRed[t + 192];
    score1[(size_t)b * NN + j0 + t] = s;
  }
}

// ---------------------------------------------------------------------------
// Kernel 3: hsum[bd] = sum_j Hbd[bd][j]
// ---------------------------------------------------------------------------
__global__ void __launch_bounds__(256) hsum_kernel(const bf16* __restrict__ Hbd,
                                                   float* __restrict__ hsum) {
  __shared__ float red[256];
  const int bd = blockIdx.x;
  const int t  = threadIdx.x;
  const bf16* row = Hbd + (size_t)bd * NN;
  float s = 0.0f;
#pragma unroll
  for (int p = 0; p < 16; ++p) s += (float)row[t + p * 256];
  red[t] = s;
  __syncthreads();
  for (int o = 128; o > 0; o >>= 1) {
    if (t < o) red[t] += red[t + o];
    __syncthreads();
  }
  if (t == 0) hsum[bd] = red[0];
}

// ---------------------------------------------------------------------------
// Kernel 4: WMMA GEMM  h2[m][i] = sum_j Hbd[m][j]*MT[i][j]  with fused
//           epilogue  score[b][i] = score1[b][i] + sum_d a2[d][i]*h2
// 128x128 tile per WG, 8 waves (4x2), each wave 2x4 16x16 WMMA tiles.
// Double-buffered LDS staging via CDNA5 async-to-LDS loads (ASYNCcnt).
// ---------------------------------------------------------------------------
__global__ void __launch_bounds__(256) gemm_score2(const bf16* __restrict__ Hbd,
                                                   const bf16* __restrict__ MT,
                                                   const float* __restrict__ a,
                                                   const float* __restrict__ score1,
                                                   float* __restrict__ score) {
  __shared__ bf16 As[2][128][64];   // [buf][m][k] 32KB
  __shared__ bf16 Bs[2][128][64];   // [buf][n][k] 32KB
  __shared__ float sc[2][128];      // per-(b,i) score2 partials
  const int m0 = blockIdx.x * 128;      // bd
  const int n0 = blockIdx.y * 128;      // i
  const int t    = threadIdx.x;
  const int lane = t & 31;
  const int wave = t >> 5;
  const int wm   = wave & 3;            // 4 waves along m
  const int wn   = wave >> 2;           // 2 waves along n
  const int hi   = lane >> 4;           // lane >= 16

  // issue 8 async 16B copies (4 As + 4 Bs) per thread for one 64-wide K stage
  auto stage = [&](int bufn, int k0) {
#pragma unroll
    for (int p = 0; p < 4; ++p) {
      int c   = t + p * 256;            // 0..1023
      int row = c >> 3;                 // 8 x 16B chunks per 128B row
      int off = (c & 7) * 8;
      uint32_t lA = (uint32_t)(uintptr_t)&As[bufn][row][off];
      uint64_t gA = (uint64_t)(uintptr_t)(Hbd + (size_t)(m0 + row) * NN + k0 + off);
      asm volatile("global_load_async_to_lds_b128 %0, %1, off"
                   :: "v"(lA), "v"(gA) : "memory");
      uint32_t lB = (uint32_t)(uintptr_t)&Bs[bufn][row][off];
      uint64_t gB = (uint64_t)(uintptr_t)(MT + (size_t)(n0 + row) * NN + k0 + off);
      asm volatile("global_load_async_to_lds_b128 %0, %1, off"
                   :: "v"(lB), "v"(gB) : "memory");
    }
  };

  v8f acc[2][4];
#pragma unroll
  for (int tm = 0; tm < 2; ++tm)
#pragma unroll
    for (int tn = 0; tn < 4; ++tn)
      acc[tm][tn] = (v8f){0.f, 0.f, 0.f, 0.f, 0.f, 0.f, 0.f, 0.f};

  sc[t >> 7][t & 127] = 0.0f;           // 256 entries

  stage(0, 0);
  for (int it = 0; it < NN / 64; ++it) {
    const int buf = it & 1;
    const bool more = (it + 1 < NN / 64);
    if (more) stage(buf ^ 1, (it + 1) * 64);   // overlap next stage with compute
    // async loads complete in order: <=8 outstanding means current stage landed
    if (more) asm volatile("s_wait_asynccnt 8" ::: "memory");
    else      asm volatile("s_wait_asynccnt 0" ::: "memory");
    __syncthreads();
#pragma unroll
    for (int kk = 0; kk < 64; kk += 32) {
      // A fragments: lanes 0-15 take K[0:16], lanes 16-31 K[16:32]
      v16bf af[2];
#pragma unroll
      for (int tm = 0; tm < 2; ++tm) {
        int row = wm * 32 + tm * 16 + (lane & 15);
        v8bf lo = *(const v8bf*)&As[buf][row][kk + hi * 16];
        v8bf hh = *(const v8bf*)&As[buf][row][kk + hi * 16 + 8];
#pragma unroll
        for (int e = 0; e < 8; ++e) { af[tm][e] = lo[e]; af[tm][e + 8] = hh[e]; }
      }
      // B fragments: lane n = col, lanes 0-15 K[0:16], lanes 16-31 K[16:32]
      v16bf bfr[4];
#pragma unroll
      for (int tn = 0; tn < 4; ++tn) {
        int col = wn * 64 + tn * 16 + (lane & 15);
        bfr[tn] = *(const v16bf*)&Bs[buf][col][kk + hi * 16];
      }
#pragma unroll
      for (int tm = 0; tm < 2; ++tm)
#pragma unroll
        for (int tn = 0; tn < 4; ++tn)
          acc[tm][tn] = __builtin_amdgcn_wmma_f32_16x16x32_bf16(
              false, af[tm], false, bfr[tn], (short)0, acc[tm][tn], false, false);
    }
    __syncthreads();   // all waves done with buf before it is refilled
  }
  // Epilogue: score2 partials; C layout: VGPR r -> M = r + 8*hi, N = lane&15.
  // For a fixed (tm,tn) all 8 rows hit the same (b,i) slot -> reduce in regs,
  // one ds_add_f32 per tile.
#pragma unroll
  for (int tm = 0; tm < 2; ++tm)
#pragma unroll
    for (int tn = 0; tn < 4; ++tn) {
      int nloc  = wn * 64 + tn * 16 + (lane & 15);
      int n     = n0 + nloc;
      int mbase = wm * 32 + tm * 16 + hi * 8;
      float s = 0.0f;
#pragma unroll
      for (int r = 0; r < 8; ++r) {
        int d = (m0 + mbase + r) & 63;
        s += acc[tm][tn][r] * a[(size_t)(FD + d) * NN + n];   // a2[d][i]
      }
      atomicAdd(&sc[mbase >> 6][nloc], s);
    }
  __syncthreads();
  // This WG exclusively owns (b in {m0/64, m0/64+1}) x (i in [n0, n0+128))
  {
    int bb = t >> 7, nn = t & 127;
    int b = (m0 >> 6) + bb;
    size_t idx = (size_t)b * NN + n0 + nn;
    score[idx] = score1[idx] + sc[bb][nn];
  }
}

// ---------------------------------------------------------------------------
// Kernel 5: out[b,i,d] = leaky(score[b,i] * hsum[b,d]), score[:,0]=0
// ---------------------------------------------------------------------------
__global__ void __launch_bounds__(256) final_kernel(const float* __restrict__ score,
                                                    const float* __restrict__ hsum,
                                                    float* __restrict__ out) {
  size_t idx = (size_t)blockIdx.x * 256 + threadIdx.x;   // 32*4096*64 total
  int d = (int)(idx & 63);
  size_t bi = idx >> 6;            // b*4096 + i
  int i = (int)(bi & (NN - 1));
  int b = (int)(bi >> 12);
  float s = (i == 0) ? 0.0f : score[bi];
  float v = s * hsum[b * FD + d];
  out[idx] = (v >= 0.0f) ? v : 0.2f * v;
}

// ---------------------------------------------------------------------------
extern "C" void kernel_launch(void* const* d_in, const int* in_sizes, int n_in,
                              void* d_out, int out_size, void* d_ws, size_t ws_size,
                              hipStream_t stream) {
  const float* inp = (const float*)d_in[0];   // (32,4096,64)
  const float* W   = (const float*)d_in[1];   // (64,64)
  const float* a   = (const float*)d_in[2];   // (128,4096)
  const int*   adj = (const int*)  d_in[3];   // (4096,4096)
  float* out = (float*)d_out;

  char* ws = (char*)d_ws;
  bf16*  Hbd    = (bf16*)(ws);                                   // 16 MB
  bf16*  MT     = (bf16*)(ws + (size_t)MDIM * NN * 2);           // 32 MB
  char*  p      = ws + (size_t)MDIM * NN * 2 + (size_t)NN * NN * 2;
  float* score1 = (float*)(p);                                   // 512 KB
  float* score  = (float*)(p + (size_t)BB * NN * 4);             // 512 KB
  float* hsum   = (float*)(p + (size_t)BB * NN * 8);             // 8 KB

  prep_mask  <<<dim3(NN / 64, NN / 64), 256, 0, stream>>>(adj, MT);
  prep_h     <<<dim3(NN / 64, BB),      256, 0, stream>>>(inp, W, a, Hbd, score1);
  hsum_kernel<<<MDIM,                   256, 0, stream>>>(Hbd, hsum);
  gemm_score2<<<dim3(MDIM / 128, NN / 128), 256, 0, stream>>>(Hbd, MT, a, score1, score);
  final_kernel<<<(BB * NN * FD) / 256,  256, 0, stream>>>(score, hsum, out);
}
